// VectorQuantizerEnt_70222715290096
// MI455X (gfx1250) — matrix operand order repaired and checked by reading, not compiled, not run on hardware
//
#include <hip/hip_runtime.h>
#include <hip/hip_bf16.h>

typedef __attribute__((ext_vector_type(16))) __bf16 bf16x16;
typedef __attribute__((ext_vector_type(8)))  float  f32x8;

#define D_DIM   1024
#define K_CB    2048
#define NROWS   32768        // B*T = 16*2048
#define M_TILE  128
#define BLOCK   256          // 8 wave32
#define LDSTR   1032         // 1024 + 8 bf16 pad (bank-conflict avoidance)

union ABFrag { bf16x16 v; unsigned u[8]; };
union Pack4  { __bf16 h[4]; uint2 q; };

// ---------------------------------------------------------------------------
// Fused GEMM (bf16 WMMA) + online softmax statistics (PASS 0) /
// diversity accumulation (PASS 1).
// Block: 128 token rows x all 2048 codebook columns, 16x16 WMMA tiles.
// B tiles staged L2 -> LDS via gfx1250 async-to-LDS engine (ASYNCcnt).
// ---------------------------------------------------------------------------
template <int PASS>
__global__ __launch_bounds__(BLOCK)
void vq_gemm_pass(const float* __restrict__ inp, const __bf16* __restrict__ cbB,
                  float* __restrict__ rowM, float* __restrict__ rowZ,
                  int* __restrict__ nnIdx, float* __restrict__ nnIdxF,
                  float* __restrict__ divSum, float* __restrict__ hclustSum)
{
    extern __shared__ __align__(16) char smem[];
    __bf16* As = (__bf16*)smem;                                // [M_TILE][LDSTR]
    __bf16* Bs = (__bf16*)(smem + (size_t)M_TILE * LDSTR * 2); // [16][LDSTR]

    const int tid  = threadIdx.x;
    const int lane = tid & 31;
    const int wave = tid >> 5;
    const int lo   = lane & 15;     // tile column N / A row M
    const int hi   = lane >> 4;     // half-wave select
    const int blockRow = blockIdx.x * M_TILE;
    const int rowBase  = blockRow + wave * 16;   // this wave's 16-row strip

    // LDS byte offset of B buffer (generic ptr low 32 bits == LDS offset)
    const unsigned bBase = (unsigned)(uintptr_t)Bs;

    // ---- stage A strip: fp32 global -> bf16 LDS (once per block) ----
    for (int i = tid; i < M_TILE * (D_DIM / 4); i += BLOCK) {
        const int r  = i >> 8;            // D_DIM/4 == 256
        const int kq = (i & 255) << 2;
        const float4 v = *(const float4*)(inp + (size_t)(blockRow + r) * D_DIM + kq);
        Pack4 pk;
        pk.h[0] = (__bf16)v.x; pk.h[1] = (__bf16)v.y;
        pk.h[2] = (__bf16)v.z; pk.h[3] = (__bf16)v.w;
        *(uint2*)(As + (size_t)r * LDSTR + kq) = pk.q;
    }

    // per-thread online state: 8 rows (M = hi*8 + r), column = lo within tile
    float mval[8], zval[8], sval[8], bval[8];
    int   bidxr[8];
    float mrow[8], zinv[8];
#pragma unroll
    for (int r = 0; r < 8; ++r) {
        mval[r] = -1e30f; zval[r] = 0.f; sval[r] = 0.f;
        bval[r] = -1e30f; bidxr[r] = 0;
        mrow[r] = 0.f;    zinv[r] = 1.f;
    }
    if (PASS == 1) {
#pragma unroll
        for (int r = 0; r < 8; ++r) {
            const int row = rowBase + hi * 8 + r;
            mrow[r] = rowM[row];
            zinv[r] = 1.0f / rowZ[row];
        }
    }

    const __bf16* aRow = As + (size_t)(wave * 16 + lo) * LDSTR;
    const __bf16* bRow = Bs + (size_t)lo * LDSTR;

    // ---- walk all 128 column tiles ----
    for (int t = 0; t < K_CB / 16; ++t) {
        const int colBase = t * 16;

        __syncthreads();   // previous tile's B fully consumed (DScnt==0 by wmma waits)

        // stage B tile: 16 bf16 codebook rows, L2 -> LDS via async engine.
        // Per-lane LDS destination address keeps the +8 element row padding.
        for (int i = tid; i < 16 * (D_DIM / 8); i += BLOCK) {   // 2048 x 16B chunks
            const int c   = i >> 7;                  // codebook row within tile
            const int kq8 = (i & 127) << 3;          // element offset (8 bf16 / chunk)
            const unsigned ldsOff = bBase + (unsigned)((c * LDSTR + kq8) * 2);
            const unsigned long long ga =
                (unsigned long long)(cbB + ((size_t)(colBase + c) << 10) + kq8);
            asm volatile("global_load_async_to_lds_b128 %0, %1, off"
                         :: "v"(ldsOff), "v"(ga) : "memory");
        }
        asm volatile("s_wait_asynccnt 0x0" ::: "memory");
        __syncthreads();

        // ---- 16x16 tile GEMM over K=1024: 32 WMMA k-steps, double-buffered ----
        auto loadFrag = [&](ABFrag& af, ABFrag& bf, int k0) {
#pragma unroll
            for (int j = 0; j < 8; ++j) {
                // A 16x32 bf16 layout: VGPR j -> K = (j<4?0:16) + hi*8 + 2*(j&3)
                const int ka = k0 + ((j & 4) << 2) + (hi << 3) + ((j & 3) << 1);
                af.u[j] = *(const unsigned*)(aRow + ka);
                // B 32x16 bf16 layout: VGPR j -> K = (j<4?0:16) + hi*8 + 2*(j&3), lane lo = N
                const int kb = k0 + ((j & 4) << 2) + (hi << 3) + ((j & 3) << 1);
                bf.u[j] = *(const unsigned*)(bRow + kb);
            }
        };

        f32x8 acc = {};
        ABFrag afr[2], bfr[2];
        loadFrag(afr[0], bfr[0], 0);
#pragma unroll 8
        for (int k0 = 0; k0 < D_DIM; k0 += 32) {
            const int cur = (k0 >> 5) & 1;
            if (k0 + 32 < D_DIM)
                loadFrag(afr[cur ^ 1], bfr[cur ^ 1], k0 + 32);   // prefetch next step
            acc = __builtin_amdgcn_wmma_f32_16x16x32_bf16(
                      false, afr[cur].v, false, bfr[cur].v, (short)0, acc, false, false);
        }

        const int col = colBase + lo;
        if (PASS == 0) {
            // online softmax + argmax update (lane-local columns)
#pragma unroll
            for (int r = 0; r < 8; ++r) {
                const float x = acc[r];
                if (x > bval[r]) { bval[r] = x; bidxr[r] = col; }
                if (x <= mval[r]) {
                    const float e = __expf(x - mval[r]);
                    zval[r] += e;
                    sval[r] += (x - mval[r]) * e;
                } else {
                    const float sc = __expf(mval[r] - x);
                    sval[r] = (sval[r] + (mval[r] - x) * zval[r]) * sc;
                    zval[r] = zval[r] * sc + 1.0f;
                    mval[r] = x;
                }
            }
        } else {
            // diversity: column sums of softmax probabilities
            float ps = 0.f;
#pragma unroll
            for (int r = 0; r < 8; ++r)
                ps += __expf(acc[r] - mrow[r]) * zinv[r];
            ps += __shfl_xor(ps, 16, 32);            // fold the two half-waves
            if (hi == 0)
                unsafeAtomicAdd(&divSum[col], ps);
        }
    }

    if (PASS == 0) {
        // merge softmax/argmax states across the 16 lanes sharing each row
#pragma unroll
        for (int r = 0; r < 8; ++r) {
            for (int msk = 1; msk < 16; msk <<= 1) {
                const float om  = __shfl_xor(mval[r],  msk, 16);
                const float oz  = __shfl_xor(zval[r],  msk, 16);
                const float os  = __shfl_xor(sval[r],  msk, 16);
                const float obv = __shfl_xor(bval[r],  msk, 16);
                const int   obi = __shfl_xor(bidxr[r], msk, 16);
                const float mN = fmaxf(mval[r], om);
                const float s1 = __expf(mval[r] - mN);
                const float s2 = __expf(om - mN);
                sval[r] = (sval[r] + (mval[r] - mN) * zval[r]) * s1
                        + (os      + (om      - mN) * oz)      * s2;
                zval[r] = zval[r] * s1 + oz * s2;
                mval[r] = mN;
                if (obv > bval[r] || (obv == bval[r] && obi < bidxr[r])) {
                    bval[r] = obv; bidxr[r] = obi;
                }
            }
        }
        if (lo == 0) {
            float hacc = 0.f;
#pragma unroll
            for (int r = 0; r < 8; ++r) {
                const int row = rowBase + hi * 8 + r;
                rowM[row]  = mval[r];
                rowZ[row]  = zval[r];
                nnIdx[row] = bidxr[r];
                nnIdxF[row] = (float)bidxr[r];
                // -sum p log2 p = (ln Z - S/Z) * log2(e)
                hacc += (__logf(zval[r]) - sval[r] / zval[r]) * 1.4426950408889634f;
            }
            unsafeAtomicAdd(hclustSum, hacc);
        }
    }
}

// ---------------------------------------------------------------------------
// Codebook fp32 -> bf16 (once; 4 MB result lives in L2 for the GEMM passes)
// ---------------------------------------------------------------------------
__global__ __launch_bounds__(256)
void vq_cvt_cb(const float* __restrict__ src, __bf16* __restrict__ dst, int nquads)
{
    const int i = blockIdx.x * 256 + threadIdx.x;
    if (i < nquads) {
        const float4 v = ((const float4*)src)[i];
        Pack4 pk;
        pk.h[0] = (__bf16)v.x; pk.h[1] = (__bf16)v.y;
        pk.h[2] = (__bf16)v.z; pk.h[3] = (__bf16)v.w;
        ((uint2*)dst)[i] = pk.q;
    }
}

// ---------------------------------------------------------------------------
// Scalar loss: h_clust - GAMMA * h_diversity
// ---------------------------------------------------------------------------
__global__ __launch_bounds__(256)
void vq_loss(const float* __restrict__ divSum, const float* __restrict__ hclustSum,
             float* __restrict__ loss)
{
    __shared__ float red[256];
    const int tid = threadIdx.x;
    float acc = 0.f;
    for (int i = tid; i < K_CB; i += 256) {
        const float d = divSum[i] * (1.0f / NROWS);
        acc += d * __log2f(d + 1e-8f);
    }
    red[tid] = acc;
    __syncthreads();
    for (int off = 128; off > 0; off >>= 1) {
        if (tid < off) red[tid] += red[tid + off];
        __syncthreads();
    }
    if (tid == 0) {
        const float h_div   = -red[0];
        const float h_clust = hclustSum[0] * (1.0f / NROWS);
        loss[0] = h_clust - 1.0f * h_div;   // GAMMA = 1.0
    }
}

// ---------------------------------------------------------------------------
// Gather chosen codebook row, mean-center, L2-normalize. One block per token.
// ---------------------------------------------------------------------------
__global__ __launch_bounds__(256)
void vq_quantize(const float* __restrict__ cb, const int* __restrict__ nnIdx,
                 float* __restrict__ out0)
{
    __shared__ float red[256];
    const int row = blockIdx.x;
    const int tid = threadIdx.x;
    const int idx = nnIdx[row];
    const float4 v = *(const float4*)(cb + (size_t)idx * D_DIM + tid * 4);

    red[tid] = v.x + v.y + v.z + v.w;
    __syncthreads();
    for (int off = 128; off > 0; off >>= 1) {
        if (tid < off) red[tid] += red[tid + off];
        __syncthreads();
    }
    const float mean = red[0] * (1.0f / D_DIM);
    __syncthreads();

    const float cx = v.x - mean, cy = v.y - mean, cz = v.z - mean, cw = v.w - mean;
    red[tid] = cx * cx + cy * cy + cz * cz + cw * cw;
    __syncthreads();
    for (int off = 128; off > 0; off >>= 1) {
        if (tid < off) red[tid] += red[tid + off];
        __syncthreads();
    }
    const float inv = rsqrtf(red[0]);
    const float4 o = {cx * inv, cy * inv, cz * inv, cw * inv};
    *(float4*)(out0 + (size_t)row * D_DIM + tid * 4) = o;
}

// ---------------------------------------------------------------------------
// Broadcast-fill quantization_loss (268 MB -> pure HBM store stream).
// ---------------------------------------------------------------------------
__global__ __launch_bounds__(256)
void vq_fill(float* __restrict__ out1, const float* __restrict__ loss, size_t nquads)
{
    const float v = *loss;
    const size_t i = (size_t)blockIdx.x * blockDim.x + threadIdx.x;
    if (i < nquads) {
        const float4 o = {v, v, v, v};
        ((float4*)out1)[i] = o;
    }
}

__global__ __launch_bounds__(256)
void vq_copy(const float* __restrict__ src, float* __restrict__ dst, size_t nquads)
{
    const size_t i = (size_t)blockIdx.x * blockDim.x + threadIdx.x;
    if (i < nquads) ((float4*)dst)[i] = ((const float4*)src)[i];
}

__global__ void vq_zero(float* __restrict__ divSum, float* __restrict__ hclustSum)
{
    const int i = blockIdx.x * blockDim.x + threadIdx.x;
    if (i < K_CB) divSum[i] = 0.f;
    if (i == 0)   hclustSum[0] = 0.f;
}

// ---------------------------------------------------------------------------
extern "C" void kernel_launch(void* const* d_in, const int* in_sizes, int n_in,
                              void* d_out, int out_size, void* d_ws, size_t ws_size,
                              hipStream_t stream)
{
    (void)in_sizes; (void)n_in; (void)out_size; (void)ws_size;

    const float* inp = (const float*)d_in[0];   // [16,2048,1024] f32
    const float* cb  = (const float*)d_in[1];   // [2048,1024]   f32

    float*  ws        = (float*)d_ws;
    __bf16* cbB       = (__bf16*)ws;            // [2048*1024] bf16 (4 MB)
    float*  rowM      = ws + 1048576;           // [32768]
    float*  rowZ      = rowM + 32768;           // [32768]
    int*    nnIdx     = (int*)(rowZ + 32768);   // [32768]
    float*  divSum    = (float*)(nnIdx + 32768);// [2048]
    float*  hclustSum = divSum + K_CB;          // [1]
    float*  lossP     = hclustSum + 1;          // [1]

    float* out0 = (float*)d_out;                         // quantized [16,2048,1024]
    float* out1 = out0 + (size_t)NROWS * D_DIM;          // quantization_loss [16,2048,2048]
    float* out2 = out1 + (size_t)NROWS * K_CB;           // nn_idx [16,2048] (as float)
    float* out3 = out2 + (size_t)NROWS;                  // codebook [2048,1024]

    const size_t shmem = (size_t)(M_TILE + 16) * LDSTR * 2;   // ~290 KB < 320 KB/WGP

    vq_zero<<<(K_CB + 255) / 256, 256, 0, stream>>>(divSum, hclustSum);

    const int cbQuads4 = K_CB * D_DIM / 4;               // 524,288
    vq_cvt_cb<<<(cbQuads4 + 255) / 256, 256, 0, stream>>>(cb, cbB, cbQuads4);

    vq_gemm_pass<0><<<NROWS / M_TILE, BLOCK, shmem, stream>>>(
        inp, cbB, rowM, rowZ, nnIdx, out2, divSum, hclustSum);
    vq_gemm_pass<1><<<NROWS / M_TILE, BLOCK, shmem, stream>>>(
        inp, cbB, rowM, rowZ, nnIdx, out2, divSum, hclustSum);

    vq_loss<<<1, 256, 0, stream>>>(divSum, hclustSum, lossP);
    vq_quantize<<<NROWS, 256, 0, stream>>>(cb, nnIdx, out0);

    const size_t fillQuads = (size_t)NROWS * K_CB / 4;   // 16,777,216
    vq_fill<<<(unsigned)((fillQuads + 255) / 256), 256, 0, stream>>>(out1, lossP, fillQuads);

    const size_t cbQuads = (size_t)K_CB * D_DIM / 4;     // 524,288
    vq_copy<<<(unsigned)((cbQuads + 255) / 256), 256, 0, stream>>>(cb, out3, cbQuads);
}
